// NVFP4Linear_68951404970073
// MI455X (gfx1250) — matrix operand order repaired and checked by reading, not compiled, not run on hardware
//
#include <hip/hip_runtime.h>

// ---------------------------------------------------------------------------
// NVFP4 fake-quant linear for MI455X (gfx1250, wave32, WMMA)
// y = x @ dequant(nvfp4(W))^T + b
// W_dq = e2m1_code * e4m3_scale  ->  <=5 significand bits -> EXACT in bf16.
// GEMM: bf16 WMMA (v_wmma_f32_16x16x32_bf16), f32 accumulate,
// async global->LDS double-buffered pipeline (ASYNCcnt).
// ---------------------------------------------------------------------------

typedef __attribute__((ext_vector_type(16))) __bf16 v16bf;
typedef __attribute__((ext_vector_type(8)))  float  v8f;
typedef __attribute__((ext_vector_type(4)))  int    v4i_a;

#if __has_builtin(__builtin_amdgcn_global_load_async_to_lds_b128) && \
    __has_builtin(__builtin_amdgcn_s_wait_asynccnt)
#define USE_ASYNC_LDS 1
typedef __attribute__((address_space(1))) v4i_a* as1_v4i_p;
typedef __attribute__((address_space(3))) v4i_a* as3_v4i_p;
#define TO_AS1(p) ((as1_v4i_p)(unsigned long long)(p))
#define TO_AS3(p) ((as3_v4i_p)(unsigned)(unsigned long long)(p))
#else
#define USE_ASYNC_LDS 0
#endif

// ------------------------------ quantization -------------------------------

__device__ __forceinline__ float nv4_code(float a) {
    // nearest value in {0,0.5,1,1.5,2,3,4,6}; ties -> lower (matches argmin)
    if (a <= 0.25f) return 0.0f;
    if (a <= 0.75f) return 0.5f;
    if (a <= 1.25f) return 1.0f;
    if (a <= 1.75f) return 1.5f;
    if (a <= 2.5f)  return 2.0f;
    if (a <= 3.5f)  return 3.0f;
    if (a <= 5.0f)  return 4.0f;
    return 6.0f;
}

__device__ __forceinline__ float e4m3_rne(float x) {
    // round-to-nearest-even f32 -> float8_e4m3fn -> f32 (x >= 0 here)
    if (!(x > 0.0f)) return 0.0f;
    if (x >= 464.0f) return 448.0f;           // saturate (never hit here)
    if (x < 0.015625f) {                      // subnormal: quantum 2^-9
        return rintf(x * 512.0f) * 0.001953125f;
    }
    unsigned u = __float_as_uint(x);
    int e = (int)(u >> 23) - 127;             // x in [2^e, 2^(e+1))
    float m = ldexpf(x, 3 - e);               // in [8,16)
    return ldexpf(rintf(m), e - 3);
}

__global__ __launch_bounds__(256) void nvfp4_quant_kernel(
    const float* __restrict__ w, __bf16* __restrict__ wq, int total_blocks) {
    int b = blockIdx.x * blockDim.x + threadIdx.x;
    if (b >= total_blocks) return;
    const float* src = w + (size_t)b * 32;

    float v[32];
#pragma unroll
    for (int i = 0; i < 8; ++i) {
        float4 t = ((const float4*)src)[i];
        v[4 * i + 0] = t.x; v[4 * i + 1] = t.y;
        v[4 * i + 2] = t.z; v[4 * i + 3] = t.w;
    }

    float amax = 0.0f;
#pragma unroll
    for (int i = 0; i < 32; ++i) amax = fmaxf(amax, fabsf(v[i]));
    amax = fmaxf(amax, 1e-12f);

    float best_mse = 3.402823466e38f;
    float best_scale = amax;
    for (int s = 0; s < 10; ++s) {
        float ratio = 0.7f + (float)s * (0.3f / 9.0f);
        float scale = amax * ratio;
        float inv = 6.0f / scale;
        float sc6 = scale * (1.0f / 6.0f);
        float mse = 0.0f;
#pragma unroll
        for (int i = 0; i < 32; ++i) {
            float sc = fminf(fmaxf(v[i] * inv, -6.0f), 6.0f);
            float q = copysignf(nv4_code(fabsf(sc)), sc);
            float d = v[i] - q * sc6;
            mse += d * d;
        }
        if (mse < best_mse) { best_mse = mse; best_scale = scale; }
    }

    float s8  = e4m3_rne(best_scale * (1.0f / 6.0f));  // fp8 ue4m3 scale
    float inv = 6.0f / fmaxf(best_scale, 1e-12f);

    union { __bf16 h[32]; uint4 u[4]; } r;
#pragma unroll
    for (int i = 0; i < 32; ++i) {
        float sc = fminf(fmaxf(v[i] * inv, -6.0f), 6.0f);
        float q = copysignf(nv4_code(fabsf(sc)), sc);
        r.h[i] = (__bf16)(q * s8);                     // exact in bf16
    }
    uint4* dst = (uint4*)(wq + (size_t)b * 32);
#pragma unroll
    for (int i = 0; i < 4; ++i) dst[i] = r.u[i];
}

// --------------------------- x : f32 -> bf16 -------------------------------

__global__ __launch_bounds__(256) void cvt_bf16_kernel(
    const float* __restrict__ x, __bf16* __restrict__ xb, size_t n8) {
    size_t i = (size_t)blockIdx.x * blockDim.x + threadIdx.x;
    if (i >= n8) return;
    const float4* p = (const float4*)x + i * 2;
    float4 a = p[0], b = p[1];
    union { __bf16 h[8]; uint4 u; } r;
    r.h[0] = (__bf16)a.x; r.h[1] = (__bf16)a.y;
    r.h[2] = (__bf16)a.z; r.h[3] = (__bf16)a.w;
    r.h[4] = (__bf16)b.x; r.h[5] = (__bf16)b.y;
    r.h[6] = (__bf16)b.z; r.h[7] = (__bf16)b.w;
    *((uint4*)xb + i) = r.u;
}

// ------------------------------- GEMM --------------------------------------

#define TILE_M 128
#define TILE_N 128
#define TILE_K 64
#define LDSS   72   // 64 + 8 pad: row*36 dwords mod 64 banks -> conflict-free

__device__ __forceinline__ v16bf load_frag_a(const __bf16* lds, int row, int kk, int lane) {
    // A 16x32 bf16: lanes 0-15 row M=l hold K kk..kk+7 & kk+16..kk+23; lanes 16-31 +8
    int l = lane & 15, hi = (lane >> 4) & 1;
    const __bf16* p = lds + (row + l) * LDSS + kk + hi * 8;
    union { uint4 u[2]; v16bf v; } f;
    f.u[0] = *(const uint4*)(p);
    f.u[1] = *(const uint4*)(p + 16);
    return f.v;
}

__device__ __forceinline__ v16bf load_frag_b(const __bf16* lds, int row, int kk, int lane) {
    // B 32x16 bf16: lanes 0-15 col N=l hold K kk..kk+7 & kk+16..kk+23; lanes 16-31 +8
    int l = lane & 15, hi = (lane >> 4) & 1;
    const __bf16* p = lds + (row + l) * LDSS + kk + hi * 8;
    union { uint4 u[2]; v16bf v; } f;
    f.u[0] = *(const uint4*)(p);
    f.u[1] = *(const uint4*)(p + 16);
    return f.v;
}

__global__ __launch_bounds__(256) void gemm_bf16_kernel(
    const __bf16* __restrict__ A,   // x_bf16  [M][K]
    const __bf16* __restrict__ B,   // w_dq    [N][K]
    const float*  __restrict__ bias,
    float* __restrict__ C,          // out [M][N]
    int M, int N, int K) {
    __shared__ __bf16 sA[2][TILE_M * LDSS];
    __shared__ __bf16 sB[2][TILE_N * LDSS];

    const int tid  = threadIdx.x;
    const int lane = tid & 31;
    const int wave = tid >> 5;
    const int bM = blockIdx.y * TILE_M;
    const int bN = blockIdx.x * TILE_N;
    const int wm = (wave >> 2) * 64;   // wave sub-tile: 64(M) x 32(N)
    const int wn = (wave & 3) * 32;

    v8f acc[4][2];
#pragma unroll
    for (int i = 0; i < 4; ++i)
#pragma unroll
        for (int j = 0; j < 2; ++j) acc[i][j] = (v8f){};

    // global->LDS copy plan: 128x64 bf16 = 1024 16B chunks, 4 per thread
    int arow[4], acol[4];
#pragma unroll
    for (int i = 0; i < 4; ++i) {
        int c = tid + i * 256;
        arow[i] = c >> 3;
        acol[i] = (c & 7) * 8;
    }

#if USE_ASYNC_LDS
    // ---- prologue: async-copy tile 0 into buffer 0 (8 instr per wave) ----
#pragma unroll
    for (int i = 0; i < 4; ++i) {
        __builtin_amdgcn_global_load_async_to_lds_b128(
            TO_AS1(A + (size_t)(bM + arow[i]) * K + acol[i]),
            TO_AS3(&sA[0][arow[i] * LDSS + acol[i]]), 0, 0);
        __builtin_amdgcn_global_load_async_to_lds_b128(
            TO_AS1(B + (size_t)(bN + arow[i]) * K + acol[i]),
            TO_AS3(&sB[0][arow[i] * LDSS + acol[i]]), 0, 0);
    }

    for (int k0 = 0; k0 < K; k0 += TILE_K) {
        const int b  = (k0 / TILE_K) & 1;
        const int kn = k0 + TILE_K;
        if (kn < K) {   // kick off next tile into the other buffer
#pragma unroll
            for (int i = 0; i < 4; ++i) {
                __builtin_amdgcn_global_load_async_to_lds_b128(
                    TO_AS1(A + (size_t)(bM + arow[i]) * K + kn + acol[i]),
                    TO_AS3(&sA[b ^ 1][arow[i] * LDSS + acol[i]]), 0, 0);
                __builtin_amdgcn_global_load_async_to_lds_b128(
                    TO_AS1(B + (size_t)(bN + arow[i]) * K + kn + acol[i]),
                    TO_AS3(&sB[b ^ 1][arow[i] * LDSS + acol[i]]), 0, 0);
            }
            __builtin_amdgcn_s_wait_asynccnt(8);  // current tile done (in-order)
        } else {
            __builtin_amdgcn_s_wait_asynccnt(0);
        }
        __syncthreads();

        const __bf16* cA = sA[b];
        const __bf16* cB = sB[b];
#pragma unroll
        for (int kk = 0; kk < TILE_K; kk += 32) {
            v16bf b0 = load_frag_b(cB, wn,      kk, lane);
            v16bf b1 = load_frag_b(cB, wn + 16, kk, lane);
#pragma unroll
            for (int i = 0; i < 4; ++i) {
                v16bf a0 = load_frag_a(cA, wm + i * 16, kk, lane);
                acc[i][0] = __builtin_amdgcn_wmma_f32_16x16x32_bf16(
                    false, a0, false, b0, (short)0, acc[i][0], false, false);
                acc[i][1] = __builtin_amdgcn_wmma_f32_16x16x32_bf16(
                    false, a0, false, b1, (short)0, acc[i][1], false, false);
            }
        }
        __syncthreads();   // all waves done reading buf b before it is refilled
    }
#else
    // ---- fallback: direct load->store copy, single buffer (sA[0]/sB[0]) ----
    for (int k0 = 0; k0 < K; k0 += TILE_K) {
#pragma unroll
        for (int i = 0; i < 4; ++i) {
            uint4 ta = *(const uint4*)(A + (size_t)(bM + arow[i]) * K + k0 + acol[i]);
            uint4 tb = *(const uint4*)(B + (size_t)(bN + arow[i]) * K + k0 + acol[i]);
            *(uint4*)(&sA[0][arow[i] * LDSS + acol[i]]) = ta;
            *(uint4*)(&sB[0][arow[i] * LDSS + acol[i]]) = tb;
        }
        __syncthreads();
#pragma unroll
        for (int kk = 0; kk < TILE_K; kk += 32) {
            v16bf b0 = load_frag_b(sB[0], wn,      kk, lane);
            v16bf b1 = load_frag_b(sB[0], wn + 16, kk, lane);
#pragma unroll
            for (int i = 0; i < 4; ++i) {
                v16bf a0 = load_frag_a(sA[0], wm + i * 16, kk, lane);
                acc[i][0] = __builtin_amdgcn_wmma_f32_16x16x32_bf16(
                    false, a0, false, b0, (short)0, acc[i][0], false, false);
                acc[i][1] = __builtin_amdgcn_wmma_f32_16x16x32_bf16(
                    false, a0, false, b1, (short)0, acc[i][1], false, false);
            }
        }
        __syncthreads();
    }
#endif

    // epilogue: C/D layout -> lanes 0-15: M=v, N=l ; lanes 16-31: M=v+8
    const int l = lane & 15, hi = lane >> 4;
#pragma unroll
    for (int i = 0; i < 4; ++i) {
#pragma unroll
        for (int j = 0; j < 2; ++j) {
            int col  = bN + wn + j * 16 + l;
            int row0 = bM + wm + i * 16 + hi * 8;
            float bv = bias[col];
#pragma unroll
            for (int v = 0; v < 8; ++v)
                C[(size_t)(row0 + v) * N + col] = acc[i][j][v] + bv;
        }
    }
}

// ------------------------------- launch ------------------------------------

extern "C" void kernel_launch(void* const* d_in, const int* in_sizes, int n_in,
                              void* d_out, int out_size, void* d_ws, size_t ws_size,
                              hipStream_t stream) {
    const float* x    = (const float*)d_in[0];   // [M][K] f32
    const float* w    = (const float*)d_in[1];   // [N][K] f32
    const float* bias = (const float*)d_in[2];   // [N]    f32
    float* out = (float*)d_out;                  // [M][N] f32

    const int N = in_sizes[2];            // 4096
    const int K = in_sizes[1] / N;        // 4096
    const int M = in_sizes[0] / K;        // 8192

    __bf16* wq = (__bf16*)d_ws;                                           // [N][K] bf16
    __bf16* xb = (__bf16*)((char*)d_ws + (size_t)N * K * sizeof(__bf16)); // [M][K] bf16

    // 1) NVFP4 fake-quant of W -> exact bf16 dequantized weight
    int total_blocks = (N * K) / 32;
    nvfp4_quant_kernel<<<(total_blocks + 255) / 256, 256, 0, stream>>>(w, wq, total_blocks);

    // 2) x -> bf16
    size_t n8 = ((size_t)M * K) / 8;
    cvt_bf16_kernel<<<(unsigned)((n8 + 255) / 256), 256, 0, stream>>>(x, xb, n8);

    // 3) bf16 WMMA GEMM + bias
    dim3 grid(N / TILE_N, M / TILE_M);
    gemm_bf16_kernel<<<grid, 256, 0, stream>>>(xb, wq, bias, out, M, N, K);
}
